// DCMoe_50972671868962
// MI455X (gfx1250) — matrix operand order repaired and checked by reading, not compiled, not run on hardware
//
#include <hip/hip_runtime.h>
#include <hip/hip_bf16.h>
#include <math.h>

#define B_TOK 2048   // BSZ*SEQ tokens
#define HDIM  1024
#define MDIM  2048
#define NEXP  8
#define TILE_T 32    // tokens per FFN tile (2 WMMA row-tiles)
#define NTILES (B_TOK / TILE_T)   // 64

typedef __attribute__((ext_vector_type(16))) __bf16 v16bf;
typedef __attribute__((ext_vector_type(8)))  float  v8f;
typedef __attribute__((ext_vector_type(8)))  unsigned v8u;

// ---------------- helpers ----------------

// pack two f32 -> bf16 pair (f0 low, f1 high): 2 adds + 1 v_perm_b32
__device__ __forceinline__ unsigned pack2bf(float f0, float f1) {
    unsigned u0 = __builtin_bit_cast(unsigned, f0) + 0x8000u;   // round-half-up
    unsigned u1 = __builtin_bit_cast(unsigned, f1) + 0x8000u;
    return __builtin_amdgcn_perm(u1, u0, 0x07060302u);          // {u1.hi16, u0.hi16}
}

__device__ __forceinline__ __bf16 f2bf(float f) {
    unsigned u = __builtin_bit_cast(unsigned, f) + 0x8000u;
    unsigned short h = (unsigned short)(u >> 16);
    __bf16 out; __builtin_memcpy(&out, &h, 2);
    return out;
}

__device__ __forceinline__ float fast_sigmoid(float x) {
    return __builtin_amdgcn_rcpf(1.f + __expf(-x));
}

// A fragment: two 16B-aligned chunks of 8 bf16 from LDS -> 2x ds_load_b128
__device__ __forceinline__ v16bf load_a_lds(const __bf16* p0, const __bf16* p1) {
    uint4 lo = *reinterpret_cast<const uint4*>(p0);
    uint4 hi = *reinterpret_cast<const uint4*>(p1);
    v8u u = { lo.x, lo.y, lo.z, lo.w, hi.x, hi.y, hi.z, hi.w };
    return __builtin_bit_cast(v16bf, u);
}

// B fragment: 16 f32 strided by ld over k (k = 0..7, 16..23 from p) from global + cvt
__device__ __forceinline__ v16bf load_b_f32(const float* p, int ld) {
    v8u u;
#pragma unroll
    for (int j = 0; j < 4; ++j) {
        u[j]     = pack2bf(p[(size_t)(2 * j) * ld], p[(size_t)(2 * j + 1) * ld]);
        u[4 + j] = pack2bf(p[(size_t)(16 + 2 * j) * ld], p[(size_t)(17 + 2 * j) * ld]);
    }
    return __builtin_bit_cast(v16bf, u);
}

// ---------------- kernel 1: zero output + counters ----------------
__global__ void moe_zero_kernel(float* out, int* counts) {
    int i = blockIdx.x * blockDim.x + threadIdx.x;
    if (i < B_TOK * HDIM) out[i] = 0.f;
    if (blockIdx.x == 0 && threadIdx.x < 16) counts[threadIdx.x] = 0;
}

// ---------------- kernel 2: routing (logits, top-2, sigmoid) ----------------
// 256 threads = 32 tokens x 8 experts
__global__ void moe_route_kernel(const float* __restrict__ x,
                                 const float* __restrict__ gate_w,
                                 const float* __restrict__ beta,
                                 int* __restrict__ topk_idx,
                                 float* __restrict__ topk_w) {
    const int lt = threadIdx.x >> 3;     // token-in-block 0..31
    const int e  = threadIdx.x & 7;      // expert 0..7
    const int t  = blockIdx.x * 32 + lt;

    const float4* xr = reinterpret_cast<const float4*>(x + (size_t)t * HDIM);
    const float4* gr = reinterpret_cast<const float4*>(gate_w + (size_t)e * HDIM);
    float acc = 0.f;
#pragma unroll 8
    for (int i = 0; i < HDIM / 4; ++i) {
        float4 a = xr[i], b = gr[i];
        acc += a.x * b.x + a.y * b.y + a.z * b.z + a.w * b.w;
    }
    __shared__ float lg[32][9];
    lg[lt][e] = acc;
    __syncthreads();

    if (threadIdx.x < 32) {
        const int l = threadIdx.x;
        const int tt = blockIdx.x * 32 + l;
        float b0 = -1e30f; int i0 = 0;
#pragma unroll
        for (int k = 0; k < NEXP; ++k) {
            float s = lg[l][k] + beta[k];
            if (s > b0) { b0 = s; i0 = k; }
        }
        float b1 = -1e30f; int i1 = 0;
#pragma unroll
        for (int k = 0; k < NEXP; ++k) {
            if (k == i0) continue;
            float s = lg[l][k] + beta[k];
            if (s > b1) { b1 = s; i1 = k; }
        }
        topk_idx[tt * 2 + 0] = i0;
        topk_idx[tt * 2 + 1] = i1;
        topk_w[tt * 2 + 0] = fast_sigmoid(lg[l][i0]);   // sigmoid of raw logit
        topk_w[tt * 2 + 1] = fast_sigmoid(lg[l][i1]);
    }
}

// ---------------- kernel 3: scatter tokens into expert buckets ----------------
__global__ void moe_scatter_kernel(const int* __restrict__ topk_idx,
                                   const float* __restrict__ topk_w,
                                   int* __restrict__ counts,
                                   int* __restrict__ tlist,
                                   float* __restrict__ twght) {
    int t = blockIdx.x * blockDim.x + threadIdx.x;
    if (t >= B_TOK) return;
#pragma unroll
    for (int j = 0; j < 2; ++j) {
        int e = topk_idx[t * 2 + j];
        int pos = atomicAdd(&counts[e], 1);
        tlist[e * B_TOK + pos] = t;
        twght[e * B_TOK + pos] = topk_w[t * 2 + j];
    }
}

// ---------------- kernel 4: expert FFN (WMMA bf16, 32-token tiles) ----------------
// grid = (NTILES, NEXP+1); block = 256 (8 waves). blockIdx.y==8 => shared expert.
__global__ __launch_bounds__(256)
void moe_ffn_kernel(const float* __restrict__ x,
                    const float* __restrict__ gate_proj_w,
                    const float* __restrict__ up_proj_w,
                    const float* __restrict__ down_proj_w,
                    const float* __restrict__ shared_gate_w,
                    const float* __restrict__ shared_up_w,
                    const float* __restrict__ shared_down_w,
                    const int* __restrict__ counts,
                    const int* __restrict__ tlist,
                    const float* __restrict__ twght,
                    float* __restrict__ out) {
    const int e    = blockIdx.y;          // 0..7 routed, 8 = shared
    const int tile = blockIdx.x;
    const int n    = (e < NEXP) ? counts[e] : B_TOK;
    const int base = tile * TILE_T;
    if (base >= n) return;                // uniform early exit (before barriers)

    const int lane = threadIdx.x & 31;
    const int wave = threadIdx.x >> 5;    // 0..7

    const float* wg = (e < NEXP) ? gate_proj_w + (size_t)e * HDIM * MDIM : shared_gate_w;
    const float* wu = (e < NEXP) ? up_proj_w   + (size_t)e * HDIM * MDIM : shared_up_w;
    const float* wd = (e < NEXP) ? down_proj_w + (size_t)e * MDIM * HDIM : shared_down_w;

    __shared__ int   stok[TILE_T];
    __shared__ float swt[TILE_T];
    __shared__ __bf16 sx[TILE_T][HDIM + 8];   // 66 KB  (row stride 1032 elem = 129*16B)
    __shared__ __bf16 sh[TILE_T][MDIM + 8];   // 132 KB

    if (threadIdx.x < TILE_T) {
        int r = threadIdx.x, idx = base + r;
        if (idx < n) {
            stok[r] = (e < NEXP) ? tlist[e * B_TOK + idx] : idx;
            swt[r]  = (e < NEXP) ? twght[e * B_TOK + idx] : 1.f;
        } else { stok[r] = 0; swt[r] = 0.f; }
    }
    __syncthreads();

    // stage x tile -> LDS (bf16 pairs). 32*512 pairs / 256 threads = 64 each
    for (int i = threadIdx.x; i < TILE_T * (HDIM / 2); i += 256) {
        int r = i >> 9;                 // / 512 pairs per row
        int h2 = (i & (HDIM / 2 - 1)) * 2;
        const float2 xv = *reinterpret_cast<const float2*>(x + (size_t)stok[r] * HDIM + h2);
        *reinterpret_cast<unsigned*>(&sx[r][h2]) = pack2bf(xv.x, xv.y);
    }
    __syncthreads();

    const int ksel  = (lane >> 4) * 8;    // A-fragment half select
    const int arow0 = lane & 15;          // row-tile 0: rows 0..15
    const int arow1 = 16 + (lane & 15);   // row-tile 1: rows 16..31

    // ---- phase 1: G = X*Wg, U = X*Wu, h = silu(G)*U  (K over H) ----
    // Each converted B fragment feeds 2 WMMAs (2 row-tiles); 4 WMMAs per k-step ({gate,up}).
    for (int mt = wave; mt < MDIM / 16; mt += 8) {
        v8f accg0 = {}, accg1 = {}, accu0 = {}, accu1 = {};
#pragma unroll 2
        for (int kt = 0; kt < HDIM / 32; ++kt) {
            const int kb = kt * 32;
            v16bf a0 = load_a_lds(&sx[arow0][kb + ksel], &sx[arow0][kb + 16 + ksel]);
            v16bf a1 = load_a_lds(&sx[arow1][kb + ksel], &sx[arow1][kb + 16 + ksel]);
            v16bf bg = load_b_f32(wg + (size_t)(kb + ksel) * MDIM + mt * 16 + (lane & 15), MDIM);
            v16bf bu = load_b_f32(wu + (size_t)(kb + ksel) * MDIM + mt * 16 + (lane & 15), MDIM);
            accg0 = __builtin_amdgcn_wmma_f32_16x16x32_bf16(false, a0, false, bg, (short)0, accg0, false, false);
            accg1 = __builtin_amdgcn_wmma_f32_16x16x32_bf16(false, a1, false, bg, (short)0, accg1, false, false);
            accu0 = __builtin_amdgcn_wmma_f32_16x16x32_bf16(false, a0, false, bu, (short)0, accu0, false, false);
            accu1 = __builtin_amdgcn_wmma_f32_16x16x32_bf16(false, a1, false, bu, (short)0, accu1, false, false);
        }
        const int dcol  = mt * 16 + (lane & 15);
        const int rbase = (lane >> 4) * 8;
#pragma unroll
        for (int v = 0; v < 8; ++v) {
            float g0 = accg0[v], u0 = accu0[v];
            float g1 = accg1[v], u1 = accu1[v];
            sh[rbase + v][dcol]      = f2bf(g0 * __builtin_amdgcn_rcpf(1.f + __expf(-g0)) * u0);
            sh[16 + rbase + v][dcol] = f2bf(g1 * __builtin_amdgcn_rcpf(1.f + __expf(-g1)) * u1);
        }
    }
    __syncthreads();

    // ---- phase 2: out_tile = h * Wd (K over M), weighted atomic combine ----
    for (int ht = wave; ht < HDIM / 16; ht += 8) {
        v8f acc0 = {}, acc1 = {};
#pragma unroll 2
        for (int kt = 0; kt < MDIM / 32; ++kt) {
            const int kb = kt * 32;
            v16bf a0 = load_a_lds(&sh[arow0][kb + ksel], &sh[arow0][kb + 16 + ksel]);
            v16bf a1 = load_a_lds(&sh[arow1][kb + ksel], &sh[arow1][kb + 16 + ksel]);
            v16bf b  = load_b_f32(wd + (size_t)(kb + ksel) * HDIM + ht * 16 + (lane & 15), HDIM);
            acc0 = __builtin_amdgcn_wmma_f32_16x16x32_bf16(false, a0, false, b, (short)0, acc0, false, false);
            acc1 = __builtin_amdgcn_wmma_f32_16x16x32_bf16(false, a1, false, b, (short)0, acc1, false, false);
        }
        const int col   = ht * 16 + (lane & 15);
        const int rbase = (lane >> 4) * 8;
#pragma unroll
        for (int v = 0; v < 8; ++v) {
            int r0 = rbase + v, r1 = 16 + rbase + v;
            if (base + r0 < n)
                atomicAdd(&out[(size_t)stok[r0] * HDIM + col], acc0[v] * swt[r0]);
            if (base + r1 < n)
                atomicAdd(&out[(size_t)stok[r1] * HDIM + col], acc1[v] * swt[r1]);
        }
    }
}

// ---------------- host launch ----------------
extern "C" void kernel_launch(void* const* d_in, const int* in_sizes, int n_in,
                              void* d_out, int out_size, void* d_ws, size_t ws_size,
                              hipStream_t stream) {
    const float* x             = (const float*)d_in[0];
    const float* gate_w        = (const float*)d_in[1];
    const float* beta          = (const float*)d_in[2];
    const float* gate_proj_w   = (const float*)d_in[3];
    const float* up_proj_w     = (const float*)d_in[4];
    const float* down_proj_w   = (const float*)d_in[5];
    const float* shared_gate_w = (const float*)d_in[6];
    const float* shared_up_w   = (const float*)d_in[7];
    const float* shared_down_w = (const float*)d_in[8];
    float* out = (float*)d_out;

    // workspace layout
    int*   counts   = (int*)d_ws;                     // 16 ints
    int*   topk_idx = counts + 16;                    // 2*B ints
    float* topk_w   = (float*)(topk_idx + 2 * B_TOK); // 2*B floats
    int*   tlist    = (int*)(topk_w + 2 * B_TOK);     // 8*B ints
    float* twght    = (float*)(tlist + NEXP * B_TOK); // 8*B floats

    moe_zero_kernel<<<(B_TOK * HDIM + 255) / 256, 256, 0, stream>>>(out, counts);
    moe_route_kernel<<<B_TOK / 32, 256, 0, stream>>>(x, gate_w, beta, topk_idx, topk_w);
    moe_scatter_kernel<<<(B_TOK + 255) / 256, 256, 0, stream>>>(topk_idx, topk_w, counts, tlist, twght);

    dim3 grid(NTILES, NEXP + 1, 1);
    moe_ffn_kernel<<<grid, 256, 0, stream>>>(x, gate_proj_w, up_proj_w, down_proj_w,
                                             shared_gate_w, shared_up_w, shared_down_w,
                                             counts, tlist, twght, out);
}
